// FeedBack_42631845380421
// MI455X (gfx1250) — matrix operand order repaired and checked by reading, not compiled, not run on hardware
//
#include <hip/hip_runtime.h>
#include <hip/hip_bf16.h>
#include <math.h>

// Problem sizes (match reference)
#define B_TOT     8192
#define T_IN      96
#define FDIM      128
#define N4        512      // 4*F
#define L_OUT     64
#define OUT_STEPS 96

typedef __attribute__((ext_vector_type(16))) __bf16 v16bf;
typedef __attribute__((ext_vector_type(8)))  __bf16 v8bf;
typedef __attribute__((ext_vector_type(8)))  float  v8f;
typedef __attribute__((ext_vector_type(4)))  float  v4f;

#define H_STRIDE 136                     // bf16 per h row (128+8 pad) -> 272B, 16B aligned
#define X_STRIDE 132                     // f32 per x row (128+4 pad)  -> 528B, 16B aligned
#define X_BUF_BYTES (32 * X_STRIDE * 4)  // 16896 B per buffer

// workspace element offsets (bf16 elements)
#define WKB_ELE 0
#define WRB_ELE 65536
#define WDB_ELE 131072
#define WSB_ELE 139264                   // Wsum = Wk + Wr (decode path)

// LDS byte offsets (dynamic LDS starts at 0: no static __shared__ in kernel)
#define LDS_WK    0
#define LDS_WR    131072
#define LDS_BIAS  262144                 // 512 f32
#define LDS_H     264192                 // 32*H_STRIDE*2 = 8704
#define LDS_X     272896                 // 2 x 16896 double buffer (warmup only)
#define LDS_WD    LDS_X                  // aliased after warmup (16384 <= 16896)
#define LDS_TOTAL (272896 + 2 * X_BUF_BYTES)   // 306688 B < 320KB WGP LDS

// ---------------------------------------------------------------------------
// Pack kernel: swizzle fp32 weights into bf16 WMMA B-fragment layout.
// B tile (32K x 16N): lane l holds n=l%16, h=l/16; elements 0..7 hold K=8h..8h+7,
// elements 8..15 hold K=16+8h..16+8h+7. Tile index = NT*4 + kt; 512 bf16 per tile.
// Also emits Wsum = Wk + Wr for the autoregressive phase (x == h).
// ---------------------------------------------------------------------------
__global__ void pack_weights(const float* __restrict__ Wk,
                             const float* __restrict__ Wr,
                             const float* __restrict__ Wd,
                             unsigned short* __restrict__ ws) {
  __bf16* o = (__bf16*)ws;
  int i = blockIdx.x * 256 + threadIdx.x;   // 544*256 = 139264 threads exactly
  if (i < 131072) {                          // WkB then WrB (65536 each)
    int j = i & 65535;
    int tile = j >> 9, le = j & 511;
    int l = le >> 4, e = le & 15;
    int NT = tile >> 2, kt = tile & 3;
    int k = kt * 32 + ((e < 8) ? 8 * (l >> 4) + e : 16 + 8 * (l >> 4) + (e - 8));
    int n = NT * 16 + (l & 15);
    int src = k * N4 + n;
    if (i < 65536) {
      float a = Wk[src], b = Wr[src];
      o[WKB_ELE + j] = (__bf16)a;
      o[WSB_ELE + j] = (__bf16)(a + b);     // fused-weight decode matrix
    } else {
      o[WRB_ELE + j] = (__bf16)Wr[src];
    }
  } else {                                   // WdB (8192)
    int j = i - 131072;
    int tile = j >> 9, le = j & 511;
    int l = le >> 4, e = le & 15;
    int NT = tile >> 2, kt = tile & 3;
    int k = kt * 32 + ((e < 8) ? 8 * (l >> 4) + e : 16 + 8 * (l >> 4) + (e - 8));
    int n = NT * 16 + (l & 15);
    o[WDB_ELE + j] = (__bf16)Wd[k * L_OUT + n];
  }
}

// ---------------------------------------------------------------------------
// device helpers
// ---------------------------------------------------------------------------
#if defined(__has_builtin)
#if __has_builtin(__builtin_amdgcn_tanhf)
#define CDNA5_HW_TANH 1
#endif
#endif

#ifdef CDNA5_HW_TANH
__device__ __forceinline__ float tanh_f(float x) { return __builtin_amdgcn_tanhf(x); }
__device__ __forceinline__ float sig_f(float x)  { return 0.5f + 0.5f * __builtin_amdgcn_tanhf(0.5f * x); }
#else
__device__ __forceinline__ float sig_f(float x)  { return 1.0f / (1.0f + __expf(-x)); }
__device__ __forceinline__ float tanh_f(float x) { return 2.0f * sig_f(2.0f * x) - 1.0f; }
#endif

__device__ __forceinline__ v8f wmma_bf16(v16bf a, v16bf b, v8f c) {
  return __builtin_amdgcn_wmma_f32_16x16x32_bf16(false, a, false, b, (short)0, c,
                                                 false, false);
}

// A fragment (16x32 bf16) for one K-tile from row-major bf16 matrix in LDS.
__device__ __forceinline__ v16bf load_a_lds(const __bf16* hrow, int lane, int kt) {
  int m = lane & 15;
  int c0 = kt * 32 + ((lane & 16) ? 8 : 0);
  const __bf16* p = hrow + m * H_STRIDE + c0;
  union { v16bf v; v8bf h[2]; } u;
  u.h[0] = *(const v8bf*)(p);        // ds_load_b128
  u.h[1] = *(const v8bf*)(p + 16);   // ds_load_b128
  return u.v;
}

// B fragment: 32 contiguous bytes per lane from the pre-swizzled tile.
__device__ __forceinline__ v16bf load_b_lds(const __bf16* wb, int tile, int lane) {
  return *(const v16bf*)(wb + tile * 512 + lane * 16);  // 2x ds_load_b128
}

#define WAIT_ASYNC(n) asm volatile("s_wait_asynccnt " #n ::: "memory")

// ---------------------------------------------------------------------------
// Persistent LSTM kernel: each block owns 32 batch rows for all 191 steps.
// 8 wave32s; wave w owns h-columns [16w,16w+16) => gate N-tiles {g*8+w}.
// Warmup x tiles are double-buffered in LDS via async global->LDS copies
// (ASYNCcnt); B fragments are software-pipelined so WMMAs don't stall on LDS.
// ---------------------------------------------------------------------------
__global__ __launch_bounds__(256)
void lstm_main(const float* __restrict__ inputs, const float* __restrict__ bias,
               const float* __restrict__ bd, const unsigned short* __restrict__ wsp,
               float* __restrict__ out) {
  extern __shared__ __align__(16) char smem[];
  __bf16* s_wk = (__bf16*)(smem + LDS_WK);   // Wk during warmup, Wsum during decode
  __bf16* s_wr = (__bf16*)(smem + LDS_WR);
  __bf16* s_wd = (__bf16*)(smem + LDS_WD);
  float*  s_b  = (float*)(smem + LDS_BIAS);
  __bf16* s_h  = (__bf16*)(smem + LDS_H);

  const int tid  = threadIdx.x;
  const int w    = tid >> 5;       // wave id 0..7
  const int lane = tid & 31;
  const int r0   = blockIdx.x * 32;
  const int tb   = w * 4;          // B-tile index base: tile(g,kt) = g*32 + tb + kt

  // Async copy of one 32x128 f32 x-tile (time step t) into LDS buffer `buf`.
  // Wave w copies rows 4w..4w+3; one b128 per lane = one full 512B row per op.
  auto issue_x_async = [&](int t, int buf) {
    const unsigned ldsbase = (unsigned)(LDS_X + buf * X_BUF_BYTES);
#pragma unroll
    for (int i = 0; i < 4; ++i) {
      const int row = w * 4 + i;
      unsigned long long ga =
          (unsigned long long)(uintptr_t)inputs +
          (((unsigned long long)(r0 + row) * T_IN + (unsigned)t) * (FDIM * 4)) +
          (unsigned)(lane * 16);
      unsigned la = ldsbase + (unsigned)(row * (X_STRIDE * 4) + lane * 16);
      asm volatile("global_load_async_to_lds_b128 %0, %1, off"
                   :: "v"(la), "v"(ga) : "memory");
    }
  };

  // kick off x tile for t=0 immediately; it overlaps the weight staging below
  issue_x_async(0, 0);

  // Stage Wk|Wr global->LDS (contiguous 262144 B = 16384 uint4)
  {
    const uint4* src = (const uint4*)wsp;
    uint4* dst = (uint4*)smem;
    for (int i = tid; i < 16384; i += 256) dst[i] = src[i];
  }
  for (int i = tid; i < 512; i += 256) s_b[i] = bias[i];
  {
    unsigned int* hz = (unsigned int*)(smem + LDS_H);
    for (int i = tid; i < (32 * H_STRIDE * 2) / 4; i += 256) hz[i] = 0u;
  }
  WAIT_ASYNC(0);
  __syncthreads();

  const int ncol = (w << 4) + (lane & 15);   // this wave's h column
  float bg[4];
#pragma unroll
  for (int g = 0; g < 4; ++g) bg[g] = s_b[g * 128 + ncol];
  const int mtP = w >> 2, ntP = w & 3;       // dense-head tile job
  const float bdv = bd[(ntP << 4) + (lane & 15)];

  v8f cst[2];
#pragma unroll
  for (int mt = 0; mt < 2; ++mt)
#pragma unroll
    for (int j = 0; j < 8; ++j) cst[mt][j] = 0.0f;

  // gate nonlinearities + c/h update + h writeback (bf16, D-layout rows)
  auto apply_gates = [&](int mt, const v8f* z) {
#pragma unroll
    for (int j = 0; j < 8; ++j) {
      float ig = sig_f(z[0][j]);
      float fg = sig_f(z[1][j]);
      float gg = tanh_f(z[2][j]);
      float og = sig_f(z[3][j]);
      float cn = fg * cst[mt][j] + ig * gg;
      cst[mt][j] = cn;
      float hn = og * tanh_f(cn);
      int m = mt * 16 + j + ((lane & 16) ? 8 : 0);
      s_h[m * H_STRIDE + ncol] = (__bf16)hn;
    }
  };

  // warmup step: z = x@Wk + h@Wr + b; depth-1 B prefetch on both matrices
  auto gates_warm = [&](int mt, const v16bf af[4], const v16bf hf[4]) {
    v8f z[4];
#pragma unroll
    for (int g = 0; g < 4; ++g)
#pragma unroll
      for (int j = 0; j < 8; ++j) z[g][j] = bg[g];

    v16bf bK = load_b_lds(s_wk, tb, lane);
    v16bf bR = load_b_lds(s_wr, tb, lane);
#pragma unroll
    for (int i = 0; i < 16; ++i) {
      const int g = i >> 2, kt = i & 3;
      v16bf cK = bK, cR = bR;
      if (i < 15) {
        const int nidx = ((i + 1) >> 2) * 32 + tb + ((i + 1) & 3);
        bK = load_b_lds(s_wk, nidx, lane);   // prefetch: completes under 2 WMMAs
        bR = load_b_lds(s_wr, nidx, lane);
      }
      z[g] = wmma_bf16(af[kt], cK, z[g]);
      z[g] = wmma_bf16(hf[kt], cR, z[g]);
    }
    apply_gates(mt, z);
  };

  // decode step: z = h@(Wk+Wr) + b (s_wk holds Wsum); depth-2 B prefetch
  auto gates_dec = [&](int mt, const v16bf hf[4]) {
    v8f z[4];
#pragma unroll
    for (int g = 0; g < 4; ++g)
#pragma unroll
      for (int j = 0; j < 8; ++j) z[g][j] = bg[g];

    v16bf b0 = load_b_lds(s_wk, tb, lane);
    v16bf b1 = load_b_lds(s_wk, tb + 1, lane);
#pragma unroll
    for (int i = 0; i < 16; ++i) {
      v16bf bc = b0;
      b0 = b1;
      if (i + 2 < 16) {
        const int nidx = ((i + 2) >> 2) * 32 + tb + ((i + 2) & 3);
        b1 = load_b_lds(s_wk, nidx, lane);   // 2 WMMAs of slack per load
      }
      z[i >> 2] = wmma_bf16(hf[i & 3], bc, z[i >> 2]);
    }
    apply_gates(mt, z);
  };

  auto do_pred = [&](int s) {   // dense head: out[:, s, :] = h @ Wd + bd
    v16bf pb[4];
#pragma unroll
    for (int kt = 0; kt < 4; ++kt)
      pb[kt] = load_b_lds(s_wd, ntP * 4 + kt, lane);   // all B up front
    v16bf pa[4];
#pragma unroll
    for (int kt = 0; kt < 4; ++kt)
      pa[kt] = load_a_lds(s_h + mtP * 16 * H_STRIDE, lane, kt);
    v8f acc;
#pragma unroll
    for (int j = 0; j < 8; ++j) acc[j] = bdv;
#pragma unroll
    for (int kt = 0; kt < 4; ++kt)
      acc = wmma_bf16(pa[kt], pb[kt], acc);
    const int n = (ntP << 4) + (lane & 15);
#pragma unroll
    for (int j = 0; j < 8; ++j) {
      int m = mtP * 16 + j + ((lane & 16) ? 8 : 0);
      out[((size_t)(r0 + m) * OUT_STEPS + s) * L_OUT + n] = acc[j];
    }
  };

  // ---------------- warmup: 96 steps over the input sequence ----------------
  for (int t = 0; t < T_IN; ++t) {
    const int buf = t & 1;
    v16bf hf[2][4];
#pragma unroll
    for (int mt = 0; mt < 2; ++mt)
#pragma unroll
      for (int kt = 0; kt < 4; ++kt)
        hf[mt][kt] = load_a_lds(s_h + mt * 16 * H_STRIDE, lane, kt);

    if (t + 1 < T_IN) {
      issue_x_async(t + 1, buf ^ 1);  // prefetch next tile into other buffer
      WAIT_ASYNC(4);                  // in-order: tile t's 4 copies are done
    } else {
      WAIT_ASYNC(0);
    }
    __syncthreads();   // h reads done before writes; x tile visible to all waves

#pragma unroll
    for (int mt = 0; mt < 2; ++mt) {
      const float* xrow = (const float*)(smem + LDS_X + buf * X_BUF_BYTES) +
                          (mt * 16 + (lane & 15)) * X_STRIDE;
      const int c0 = (lane & 16) ? 8 : 0;
      v16bf xf[4];
#pragma unroll
      for (int kt = 0; kt < 4; ++kt) {
        const float* p = xrow + kt * 32 + c0;
        v4f a0 = *(const v4f*)(p);
        v4f a1 = *(const v4f*)(p + 4);
        v4f a2 = *(const v4f*)(p + 16);
        v4f a3 = *(const v4f*)(p + 20);
        v16bf v;
#pragma unroll
        for (int q = 0; q < 4; ++q) {
          v[q]      = (__bf16)a0[q];
          v[4 + q]  = (__bf16)a1[q];
          v[8 + q]  = (__bf16)a2[q];
          v[12 + q] = (__bf16)a3[q];
        }
        xf[kt] = v;
      }
      gates_warm(mt, xf, hf[mt]);
    }
    __syncthreads();   // new h visible
  }

  // ------- transition: s_wk <- Wsum (decode GEMM), x-buffer <- Wd tiles -------
  {
    const uint4* wsum = (const uint4*)((const char*)wsp + 2 * WSB_ELE);
    uint4* dwk = (uint4*)(smem + LDS_WK);
    for (int i = tid; i < 8192; i += 256) dwk[i] = wsum[i];
    const uint4* wdsrc = (const uint4*)((const char*)wsp + 2 * WDB_ELE);
    uint4* dwd = (uint4*)(smem + LDS_WD);
    for (int i = tid; i < 1024; i += 256) dwd[i] = wdsrc[i];
  }
  __syncthreads();

  do_pred(0);          // first prediction from warmup-final h

  // ---------------- autoregressive decode: x_t = h_{t-1} ----------------
  for (int s = 1; s < OUT_STEPS; ++s) {
    v16bf hf[2][4];
#pragma unroll
    for (int mt = 0; mt < 2; ++mt)
#pragma unroll
      for (int kt = 0; kt < 4; ++kt)
        hf[mt][kt] = load_a_lds(s_h + mt * 16 * H_STRIDE, lane, kt);
    __syncthreads();

#pragma unroll
    for (int mt = 0; mt < 2; ++mt)
      gates_dec(mt, hf[mt]);   // z = h@(Wk+Wr) + b
    __syncthreads();

    do_pred(s);
  }
}

// ---------------------------------------------------------------------------
extern "C" void kernel_launch(void* const* d_in, const int* in_sizes, int n_in,
                              void* d_out, int out_size, void* d_ws, size_t ws_size,
                              hipStream_t stream) {
  const float* inputs = (const float*)d_in[0];
  const float* Wk     = (const float*)d_in[1];
  const float* Wr     = (const float*)d_in[2];
  const float* b      = (const float*)d_in[3];
  const float* Wd     = (const float*)d_in[4];
  const float* bd     = (const float*)d_in[5];
  float* out = (float*)d_out;
  unsigned short* wsb = (unsigned short*)d_ws;

  (void)hipFuncSetAttribute(reinterpret_cast<const void*>(lstm_main),
                            hipFuncAttributeMaxDynamicSharedMemorySize, LDS_TOTAL);

  pack_weights<<<544, 256, 0, stream>>>(Wk, Wr, Wd, wsb);
  lstm_main<<<B_TOT / 32, 256, LDS_TOTAL, stream>>>(inputs, b, bd, wsb, out);
}
